// TransformerBlock_29446295781404
// MI455X (gfx1250) — compile-verified, matrix operand order checked by reading, not run on hardware
//
#include <hip/hip_runtime.h>
#include <hip/hip_bf16.h>
#include <math.h>

// ---------------------------------------------------------------------------
// Types / helpers
// ---------------------------------------------------------------------------
typedef __attribute__((ext_vector_type(16))) __bf16 v16bf;
typedef __attribute__((ext_vector_type(8)))  __bf16 v8bf;
typedef __attribute__((ext_vector_type(8)))  float  v8f;

__device__ __forceinline__ __bf16 f2bf(float f) {
  unsigned u = __builtin_bit_cast(unsigned, f);
  unsigned r = u + 0x7FFFu + ((u >> 16) & 1u);   // round-to-nearest-even
  unsigned short h = (unsigned short)(r >> 16);
  return __builtin_bit_cast(__bf16, h);
}

__device__ __forceinline__ v8f zero8() {
  v8f z = {0.f, 0.f, 0.f, 0.f, 0.f, 0.f, 0.f, 0.f};
  return z;
}

__device__ __forceinline__ v8f wmma_bf16(v16bf a, v16bf b, v8f c) {
  // D(f32 16x16) = A(bf16 16x32) * B(bf16 32x16) + C
  return __builtin_amdgcn_wmma_f32_16x16x32_bf16(
      /*neg_a=*/false, a, /*neg_b=*/false, b,
      /*c_mod=*/(short)0, c, /*reuse_a=*/false, /*reuse_b=*/false);
}

// 16x32 bf16 fragment from 16 consecutive *rows* of a row-major matrix.
// Per ISA 16-bit A layout: lane r (0..15), half h: elems [0..7] = row r,
// K = k0 + h*8 .. +7 ; elems [8..15] = row r, K = k0 + 16 + h*8 .. +7.
// Identical layout serves B of X@W^T (B[k][n] = W[n][k] -> rows of W).
// Works for both global and LDS (generic) pointers.
__device__ __forceinline__ v16bf load_frag_rows(const __bf16* __restrict__ base,
                                                int ld, int row0, int k0, int lane) {
  int r = lane & 15, h = lane >> 4;
  const __bf16* p = base + (size_t)(row0 + r) * ld + (k0 + h * 8);
  v8bf lo = *(const v8bf*)p;          // 16-byte load
  v8bf hi = *(const v8bf*)(p + 16);   // 16-byte load
  v16bf f;
#pragma unroll
  for (int i = 0; i < 8; ++i) { f[i] = lo[i]; f[i + 8] = hi[i]; }
  return f;
}

// Fragment whose K dimension runs *down a column* of a row-major matrix
// (B operand of P@V: B[k][n] = V[k][n]).
__device__ __forceinline__ v16bf load_frag_cols(const __bf16* __restrict__ base,
                                                int ld, int k0, int col0, int lane) {
  int r = lane & 15, h = lane >> 4;
  const __bf16* p = base + (size_t)(k0 + h * 8) * ld + col0 + r;
  v16bf f;
#pragma unroll
  for (int i = 0; i < 8; ++i) f[i] = p[(size_t)i * ld];
  p += (size_t)16 * ld;
#pragma unroll
  for (int i = 0; i < 8; ++i) f[i + 8] = p[(size_t)i * ld];
  return f;
}

// Async Global -> LDS copy of 16 bytes (tracked by ASYNCcnt).
// lds_off: LDS byte address (low 32 bits of the generic shared pointer ==
// LDS offset per the flat-aperture mapping, ISA 10.2).
__device__ __forceinline__ void async_b128(unsigned lds_off, const void* g) {
  unsigned long long ga = (unsigned long long)g;
  asm volatile("global_load_async_to_lds_b128 %0, %1, off"
               :: "v"(lds_off), "v"(ga) : "memory");
}

// ---------------------------------------------------------------------------
// fp32 -> bf16 conversion
// ---------------------------------------------------------------------------
__global__ __launch_bounds__(256) void cvt_bf16_k(const float* __restrict__ in,
                                                  __bf16* __restrict__ out, int n) {
  int stride = gridDim.x * 256;
  for (int i = blockIdx.x * 256 + threadIdx.x; i < n; i += stride)
    out[i] = f2bf(in[i]);
}

// ---------------------------------------------------------------------------
// Y[M,N] = X[M,K] @ W[N,K]^T  (+ epilogue), bf16 WMMA, fp32 accumulate.
// 256 threads = 8 waves (4x2). Wave tile 32x32, block tile 128x64.
// Double-buffered async Global->LDS staging: per K=32 step each thread issues
// three global_load_async_to_lds_b128 (X tile 8KB, W tile 4KB), overlapped
// with WMMA compute on the previous stage via s_wait_asynccnt.
// EPI: 0 = plain -> bf16 out
//      1 = +bias, exact GELU -> bf16 out
//      2 = (+bias) + fp32 residual -> fp32 out
// ---------------------------------------------------------------------------
template <int EPI>
__global__ __launch_bounds__(256) void gemm_bf16_k(
    const __bf16* __restrict__ X, const __bf16* __restrict__ W,
    const float* __restrict__ bias, const float* __restrict__ resid,
    float* __restrict__ outF, __bf16* __restrict__ outB,
    int M, int N, int K) {
  __shared__ __attribute__((aligned(16))) __bf16 xs[2][128 * 32];
  __shared__ __attribute__((aligned(16))) __bf16 wsb[2][64 * 32];

  int tid = threadIdx.x;
  int lane = tid & 31;
  int wave = tid >> 5;
  int wm = wave & 3, wn = wave >> 4 ? 0 : (wave >> 2);  // (kept simple below)
  wn = wave >> 2;
  int m0 = wm * 32;            // within block tile
  int n0 = wn * 32;
  int mblk = blockIdx.x * 128;
  int nblk = blockIdx.y * 64;

  // Issue one K=32 stage of async copies into buffer `buf`.
  auto stage = [&](int buf, int k0) {
    // X tile: 128 rows x 32 bf16 = 512 16B-chunks; thread t -> chunks t, t+256
    int c = tid;
    int row = c >> 2, q = c & 3;
    async_b128((unsigned)(size_t)&xs[buf][row * 32 + q * 8],
               X + (size_t)(mblk + row) * K + k0 + q * 8);
    c = tid + 256;
    row = c >> 2; q = c & 3;
    async_b128((unsigned)(size_t)&xs[buf][row * 32 + q * 8],
               X + (size_t)(mblk + row) * K + k0 + q * 8);
    // W tile: 64 rows x 32 bf16 = 256 16B-chunks; thread t -> chunk t
    row = tid >> 2; q = tid & 3;
    async_b128((unsigned)(size_t)&wsb[buf][row * 32 + q * 8],
               W + (size_t)(nblk + row) * K + k0 + q * 8);
  };

  v8f acc00 = zero8(), acc01 = zero8(), acc10 = zero8(), acc11 = zero8();

  int nk = K >> 5;
  stage(0, 0);
  for (int kk = 0; kk < nk; ++kk) {
    __syncthreads();                        // buf[(kk+1)&1] free for overwrite
    if (kk + 1 < nk) {
      stage((kk + 1) & 1, (kk + 1) << 5);
      asm volatile("s_wait_asynccnt 0x3" ::: "memory");  // stage kk complete
    } else {
      asm volatile("s_wait_asynccnt 0x0" ::: "memory");
    }
    __syncthreads();                        // stage kk visible to all waves

    const __bf16* xt = xs[kk & 1];
    const __bf16* wt = wsb[kk & 1];
    v16bf a0 = load_frag_rows(xt, 32, m0, 0, lane);
    v16bf a1 = load_frag_rows(xt, 32, m0 + 16, 0, lane);
    v16bf b0 = load_frag_rows(wt, 32, n0, 0, lane);
    v16bf b1 = load_frag_rows(wt, 32, n0 + 16, 0, lane);
    acc00 = wmma_bf16(a0, b0, acc00);
    acc01 = wmma_bf16(a0, b1, acc01);
    acc10 = wmma_bf16(a1, b0, acc10);
    acc11 = wmma_bf16(a1, b1, acc11);
  }

  int r = lane & 15, h = lane >> 4;
#pragma unroll
  for (int ti = 0; ti < 2; ++ti) {
#pragma unroll
    for (int tj = 0; tj < 2; ++tj) {
      v8f acc = (ti == 0) ? (tj == 0 ? acc00 : acc01)
                          : (tj == 0 ? acc10 : acc11);
      int col = nblk + n0 + tj * 16 + r;
      float bv = (EPI == 0) ? 0.f : (bias ? bias[col] : 0.f);
#pragma unroll
      for (int i = 0; i < 8; ++i) {
        int row = mblk + m0 + ti * 16 + i + 8 * h;  // C layout: vgpr i -> M=i+8*h
        float v = acc[i] + bv;
        size_t idx = (size_t)row * N + col;
        if (EPI == 1) {
          v = 0.5f * v * (1.0f + erff(v * 0.70710678118654752f));  // exact GELU
          outB[idx] = f2bf(v);
        } else if (EPI == 2) {
          outF[idx] = v + resid[idx];
        } else {
          outB[idx] = f2bf(v);
        }
      }
    }
  }
}

// ---------------------------------------------------------------------------
// Flash-style causal attention, hd = 64, wave32.
// One wave = 16 query rows; block = 8 waves = 128 rows.
// Grid: (S/128, H, B). Q/K/V/O are [B*S, D] bf16, head h at cols h*64.
// ---------------------------------------------------------------------------
__global__ __launch_bounds__(256) void attention_k(
    const __bf16* __restrict__ Q, const __bf16* __restrict__ Km,
    const __bf16* __restrict__ V, __bf16* __restrict__ O,
    int S, int Dm) {
  __shared__ __attribute__((aligned(16))) __bf16 Plds[8 * 16 * 32];

  int lane = threadIdx.x & 31, wave = threadIdx.x >> 5;
  int r = lane & 15, hl = lane >> 4;
  int h = blockIdx.y, b = blockIdx.z;
  int qbase = blockIdx.x * 128 + wave * 16;   // wave-uniform

  size_t base = (size_t)b * S * Dm + (size_t)h * 64;
  const __bf16* qp = Q + base;
  const __bf16* kp = Km + base;
  const __bf16* vp = V + base;
  __bf16* op = O + base;
  __bf16* pl = &Plds[wave * 512];             // per-wave 16x32 staging tile

  v16bf aq0 = load_frag_rows(qp, Dm, qbase, 0, lane);
  v16bf aq1 = load_frag_rows(qp, Dm, qbase, 32, lane);

  v8f o0 = zero8(), o1 = zero8(), o2 = zero8(), o3 = zero8();
  float m[8], l[8];
#pragma unroll
  for (int i = 0; i < 8; ++i) { m[i] = -1e30f; l[i] = 0.f; }

  for (int j0 = 0; j0 < qbase + 16; j0 += 32) {   // wave-uniform trip count
    // ---- S = Q @ K^T (two 16x16 score tiles: keys j0.. and j0+16..) ----
    v8f s0 = zero8(), s1 = zero8();
    {
      v16bf bk0 = load_frag_rows(kp, Dm, j0, 0, lane);
      v16bf bk1 = load_frag_rows(kp, Dm, j0, 32, lane);
      s0 = wmma_bf16(aq0, bk0, s0);
      s0 = wmma_bf16(aq1, bk1, s0);
      v16bf bk2 = load_frag_rows(kp, Dm, j0 + 16, 0, lane);
      v16bf bk3 = load_frag_rows(kp, Dm, j0 + 16, 32, lane);
      s1 = wmma_bf16(aq0, bk2, s1);
      s1 = wmma_bf16(aq1, bk3, s1);
    }
    // ---- scale, causal mask, online softmax ----
#pragma unroll
    for (int i = 0; i < 8; ++i) {
      int qrow = qbase + i + 8 * hl;
      float sc0 = s0[i] * 0.125f;               // 1/sqrt(64)
      float sc1 = s1[i] * 0.125f;
      if (j0 + r > qrow) sc0 = -1e30f;
      if (j0 + 16 + r > qrow) sc1 = -1e30f;
      float rm = fmaxf(sc0, sc1);
#pragma unroll
      for (int off = 1; off < 16; off <<= 1) rm = fmaxf(rm, __shfl_xor(rm, off, 32));
      float mn = fmaxf(m[i], rm);
      float al = __expf(m[i] - mn);
      float p0 = __expf(sc0 - mn);
      float p1 = __expf(sc1 - mn);
      float rs = p0 + p1;
#pragma unroll
      for (int off = 1; off < 16; off <<= 1) rs += __shfl_xor(rs, off, 32);
      l[i] = l[i] * al + rs;
      m[i] = mn;
      o0[i] *= al; o1[i] *= al; o2[i] *= al; o3[i] *= al;
      // stage P (C layout -> LDS row-major 16x32, bf16)
      pl[(i + 8 * hl) * 32 + r]      = f2bf(p0);
      pl[(i + 8 * hl) * 32 + 16 + r] = f2bf(p1);
    }
    asm volatile("s_wait_dscnt 0" ::: "memory");
    // ---- O += P @ V ----
    v16bf ap  = load_frag_rows(pl, 32, 0, 0, lane);   // A-layout reload from LDS
    v16bf bv0 = load_frag_cols(vp, Dm, j0, 0, lane);
    v16bf bv1 = load_frag_cols(vp, Dm, j0, 16, lane);
    v16bf bv2 = load_frag_cols(vp, Dm, j0, 32, lane);
    v16bf bv3 = load_frag_cols(vp, Dm, j0, 48, lane);
    o0 = wmma_bf16(ap, bv0, o0);
    o1 = wmma_bf16(ap, bv1, o1);
    o2 = wmma_bf16(ap, bv2, o2);
    o3 = wmma_bf16(ap, bv3, o3);
  }

#pragma unroll
  for (int i = 0; i < 8; ++i) {
    float inv = 1.0f / l[i];
    size_t row = (size_t)(qbase + i + 8 * hl);
    op[row * Dm + 0 + r]  = f2bf(o0[i] * inv);
    op[row * Dm + 16 + r] = f2bf(o1[i] * inv);
    op[row * Dm + 32 + r] = f2bf(o2[i] * inv);
    op[row * Dm + 48 + r] = f2bf(o3[i] * inv);
  }
}

// ---------------------------------------------------------------------------
// LayerNorm over last dim (block per row). Writes fp32 (+ optional bf16).
// ---------------------------------------------------------------------------
__global__ __launch_bounds__(256) void layernorm_k(
    const float* __restrict__ in, const float* __restrict__ gamma,
    const float* __restrict__ beta, float* __restrict__ outF,
    __bf16* outB, int D) {
  __shared__ float red[256];
  int row = blockIdx.x, tid = threadIdx.x;
  const float* p = in + (size_t)row * D;

  float s = 0.f;
  for (int i = tid; i < D; i += 256) s += p[i];
  red[tid] = s; __syncthreads();
  for (int o = 128; o > 0; o >>= 1) { if (tid < o) red[tid] += red[tid + o]; __syncthreads(); }
  float mu = red[0] / (float)D;
  __syncthreads();

  float ss = 0.f;
  for (int i = tid; i < D; i += 256) { float d = p[i] - mu; ss += d * d; }
  red[tid] = ss; __syncthreads();
  for (int o = 128; o > 0; o >>= 1) { if (tid < o) red[tid] += red[tid + o]; __syncthreads(); }
  float inv = rsqrtf(red[0] / (float)D + 1e-5f);

  for (int i = tid; i < D; i += 256) {
    float y = (p[i] - mu) * inv * gamma[i] + beta[i];
    outF[(size_t)row * D + i] = y;
    if (outB) outB[(size_t)row * D + i] = f2bf(y);
  }
}

// ---------------------------------------------------------------------------
// Host orchestration
// ---------------------------------------------------------------------------
extern "C" void kernel_launch(void* const* d_in, const int* in_sizes, int n_in,
                              void* d_out, int out_size, void* d_ws, size_t ws_size,
                              hipStream_t stream) {
  (void)in_sizes; (void)n_in; (void)out_size; (void)ws_size;
  constexpr int B = 2, S = 2048, D = 1024, H = 16, M = B * S, DF = 4 * D;

  const float* x   = (const float*)d_in[0];
  const float* wq  = (const float*)d_in[1];
  const float* wk  = (const float*)d_in[2];
  const float* wv  = (const float*)d_in[3];
  const float* wo  = (const float*)d_in[4];
  const float* w1  = (const float*)d_in[5];
  const float* b1  = (const float*)d_in[6];
  const float* w2  = (const float*)d_in[7];
  const float* b2  = (const float*)d_in[8];
  const float* g0  = (const float*)d_in[9];
  const float* be0 = (const float*)d_in[10];
  const float* g1  = (const float*)d_in[11];
  const float* be1 = (const float*)d_in[12];

  char* ws = (char*)d_ws;
  size_t off = 0;
  auto take = [&](size_t bytes) -> void* {
    void* p = ws + off;
    off = (off + bytes + 255) & ~(size_t)255;
    return p;
  };

  __bf16* xb   = (__bf16*)take((size_t)M * D * 2);    // also reused as ln0b
  __bf16* wqb  = (__bf16*)take((size_t)D * D * 2);
  __bf16* wkb  = (__bf16*)take((size_t)D * D * 2);
  __bf16* wvb  = (__bf16*)take((size_t)D * D * 2);
  __bf16* wob  = (__bf16*)take((size_t)D * D * 2);
  __bf16* w1b  = (__bf16*)take((size_t)DF * D * 2);
  __bf16* w2b  = (__bf16*)take((size_t)D * DF * 2);
  size_t qkv_off = off;
  __bf16* qb    = (__bf16*)take((size_t)M * D * 2);
  __bf16* kb    = (__bf16*)take((size_t)M * D * 2);
  __bf16* vb    = (__bf16*)take((size_t)M * D * 2);
  __bf16* attnb = (__bf16*)take((size_t)M * D * 2);
  float*  x1    = (float*)take((size_t)M * D * 4);    // also reused as x2
  float*  ln0f  = (float*)take((size_t)M * D * 4);
  // reuse dead regions:
  __bf16* hb   = (__bf16*)(ws + qkv_off);             // 32MB over q/k/v/attn (dead)
  __bf16* ln0b = xb;                                  // xb dead after QKV gemms
  float*  x2   = x1;                                  // x1 dead after LN0

  dim3 blk(256);

  // fp32 -> bf16 conversions
  cvt_bf16_k<<<1024, blk, 0, stream>>>(x,  xb,  M * D);
  cvt_bf16_k<<<1024, blk, 0, stream>>>(wq, wqb, D * D);
  cvt_bf16_k<<<1024, blk, 0, stream>>>(wk, wkb, D * D);
  cvt_bf16_k<<<1024, blk, 0, stream>>>(wv, wvb, D * D);
  cvt_bf16_k<<<1024, blk, 0, stream>>>(wo, wob, D * D);
  cvt_bf16_k<<<1024, blk, 0, stream>>>(w1, w1b, DF * D);
  cvt_bf16_k<<<1024, blk, 0, stream>>>(w2, w2b, D * DF);

  // Q, K, V projections (bf16 out)
  dim3 gD(M / 128, D / 64);
  gemm_bf16_k<0><<<gD, blk, 0, stream>>>(xb, wqb, nullptr, nullptr, nullptr, qb, M, D, D);
  gemm_bf16_k<0><<<gD, blk, 0, stream>>>(xb, wkb, nullptr, nullptr, nullptr, kb, M, D, D);
  gemm_bf16_k<0><<<gD, blk, 0, stream>>>(xb, wvb, nullptr, nullptr, nullptr, vb, M, D, D);

  // causal flash attention
  attention_k<<<dim3(S / 128, H, B), blk, 0, stream>>>(qb, kb, vb, attnb, S, D);

  // output projection + residual (fp32)
  gemm_bf16_k<2><<<gD, blk, 0, stream>>>(attnb, wob, nullptr, x, x1, nullptr, M, D, D);

  // LayerNorm 0 (fp32 + bf16 copies)
  layernorm_k<<<M, blk, 0, stream>>>(x1, g0, be0, ln0f, ln0b, D);

  // MLP up + exact GELU (bf16 out)
  gemm_bf16_k<1><<<dim3(M / 128, DF / 64), blk, 0, stream>>>(ln0b, w1b, b1, nullptr, nullptr, hb, M, DF, D);

  // MLP down + bias + residual (fp32)
  gemm_bf16_k<2><<<gD, blk, 0, stream>>>(hb, w2b, b2, ln0f, x2, nullptr, M, D, DF);

  // LayerNorm 1 -> final output (fp32)
  layernorm_k<<<M, blk, 0, stream>>>(x2, g1, be1, (float*)d_out, nullptr, D);
}
